// SmoothQuantSegmentLinear_80762565034003
// MI455X (gfx1250) — 8-bit matrix operand order repaired and checked by reading, compiled, not run on hardware
//
#include <hip/hip_runtime.h>
#include <stdint.h>


typedef __attribute__((ext_vector_type(8))) int   v8i;
typedef __attribute__((ext_vector_type(4))) float v4f;

#define CHUNKS    4
#define IN_F      4096
#define OUT_F     16384
#define NTOK      8192
#define OC        (OUT_F / CHUNKS)
#define QMAXF     127.0f

#define BM        128
#define BN        128
#define BK        64
#define THREADS   256

static __device__ __forceinline__ int8_t quant1(float v, float inv_scale) {
    float q = rintf(v * inv_scale);
    q = fminf(fmaxf(q, -QMAXF), QMAXF);
    return (int8_t)(int)q;
}

static __device__ __forceinline__ void async_cp16(const int8_t* sbase,
                                                  unsigned voff, int8_t* l) {
    unsigned loff = (unsigned)(uintptr_t)l;
    asm volatile("global_load_async_to_lds_b128 %0, %1, %2"
                 :: "v"(loff), "v"(voff), "s"(sbase) : "memory");
}
static __device__ __forceinline__ void wait_async0() {
    asm volatile("s_wait_asynccnt 0x0" ::: "memory");
}

__global__ __launch_bounds__(256) void sq_quant_w(
        const float* __restrict__ w, const float* __restrict__ smooth,
        const float* __restrict__ w_scale, int8_t* __restrict__ wq) {
    size_t idx = ((size_t)blockIdx.x * blockDim.x + threadIdx.x) * 4;
    int    i   = (int)(idx & (IN_F - 1));
    size_t row = idx >> 12;
    int    k   = (int)(row >> 12);
    float  inv = 1.0f / w_scale[k];
    v4f    wv  = __builtin_nontemporal_load((const v4f*)(w + idx));
    v4f    sv  = *(const v4f*)(smooth + (size_t)k * IN_F + i);
    char4  q;
    q.x = quant1(wv.x * sv.x, inv);
    q.y = quant1(wv.y * sv.y, inv);
    q.z = quant1(wv.z * sv.z, inv);
    q.w = quant1(wv.w * sv.w, inv);
    *(char4*)(wq + idx) = q;
}

__global__ __launch_bounds__(256) void sq_quant_x(
        const float* __restrict__ x, const float* __restrict__ smooth,
        const float* __restrict__ in_scale, int8_t* __restrict__ xq) {
    size_t idx = ((size_t)blockIdx.x * blockDim.x + threadIdx.x) * 4;
    int    i   = (int)(idx & (IN_F - 1));
    size_t kb  = idx >> 12;
    int    k   = (int)(kb >> 13);
    size_t b   = kb & (NTOK - 1);
    float  s   = in_scale[k];
    v4f    xv  = *(const v4f*)(x + b * IN_F + i);
    v4f    sm  = *(const v4f*)(smooth + (size_t)k * IN_F + i);
    char4  q;
    q.x = quant1(xv.x, 1.0f / (sm.x * s));
    q.y = quant1(xv.y, 1.0f / (sm.y * s));
    q.z = quant1(xv.z, 1.0f / (sm.z * s));
    q.w = quant1(xv.w, 1.0f / (sm.w * s));
    *(char4*)(xq + idx) = q;
}

__global__ __launch_bounds__(THREADS) void sq_gemm_i8(
        const int8_t* __restrict__ xq, const int8_t* __restrict__ wq,
        const float* __restrict__ bias, const float* __restrict__ in_scale,
        const float* __restrict__ w_scale, float* __restrict__ out) {
    __shared__ int8_t lds_a[2][BM * BK];
    __shared__ int8_t lds_b[2][BN * BK];

    const int tid  = threadIdx.x;
    const int lane = tid & 31;
    const int wid  = tid >> 5;
    const int wm   = wid & 3;
    const int wn   = wid >> 2;
    const int bm   = blockIdx.x * BM;
    const int bn   = blockIdx.y * BN;
    const int kch  = bn / OC;

    const int8_t* Abase = xq + ((size_t)kch * NTOK + bm) * IN_F;
    const int8_t* Bbase = wq + (size_t)bn * IN_F;

    const int      r0    = tid >> 2;
    const int      c0    = (tid & 3) * 16;
    const unsigned voff0 = (unsigned)(r0 * IN_F + c0);
    const unsigned voff1 = (unsigned)((r0 + 64) * IN_F + c0);

    v8i acc[2][4];
#pragma unroll
    for (int mt = 0; mt < 2; ++mt)
#pragma unroll
        for (int nt = 0; nt < 4; ++nt) { v8i z = {}; acc[mt][nt] = z; }

    const int rlane = lane & 15;
    const int half  = lane >> 4;

    auto stage = [&](int buf, int k0) {
        const int8_t* sa = Abase + k0;
        const int8_t* sb = Bbase + k0;
        async_cp16(sa, voff0, &lds_a[buf][r0 * BK + c0]);
        async_cp16(sa, voff1, &lds_a[buf][(r0 + 64) * BK + c0]);
        async_cp16(sb, voff0, &lds_b[buf][r0 * BK + c0]);
        async_cp16(sb, voff1, &lds_b[buf][(r0 + 64) * BK + c0]);
    };

    auto compute = [&](int buf) {
        v8i afrag[2];
#pragma unroll
        for (int mt = 0; mt < 2; ++mt) {
            const int* rowp = (const int*)(lds_a[buf] + (wm * 32 + mt * 16 + rlane) * BK);
            v8i a;
#pragma unroll
            for (int c = 0; c < 4; ++c) {
                a[c]     = rowp[half * 4 + c];
                a[c + 4] = rowp[8 + half * 4 + c];
            }
            afrag[mt] = a;
        }
        v8i bfrag[4];
#pragma unroll
        for (int nt = 0; nt < 4; ++nt) {
            const int* rowp = (const int*)(lds_b[buf] + (wn * 64 + nt * 16 + rlane) * BK);
            int4 lo = *(const int4*)(rowp + half * 4);
            int4 hi = *(const int4*)(rowp + 8 + half * 4);
            v8i b;
            b[0] = lo.x; b[1] = lo.y; b[2] = lo.z; b[3] = lo.w;
            b[4] = hi.x; b[5] = hi.y; b[6] = hi.z; b[7] = hi.w;
            bfrag[nt] = b;
        }
#pragma unroll
        for (int mt = 0; mt < 2; ++mt)
#pragma unroll
            for (int nt = 0; nt < 4; ++nt)
                acc[mt][nt] = __builtin_amdgcn_wmma_i32_16x16x64_iu8(
                    true, afrag[mt], true, bfrag[nt], acc[mt][nt], false, false);
    };

    stage(0, 0);
    for (int k0 = 0; k0 < IN_F; k0 += 2 * BK) {
        wait_async0();
        __syncthreads();
        if (k0 + BK < IN_F) stage(1, k0 + BK);
        compute(0);

        wait_async0();
        __syncthreads();
        if (k0 + 2 * BK < IN_F) stage(0, k0 + 2 * BK);
        compute(1);
    }

    const float s = in_scale[kch] * w_scale[kch];
#pragma unroll
    for (int mt = 0; mt < 2; ++mt) {
#pragma unroll
        for (int nt = 0; nt < 4; ++nt) {
            int   col = bn + wn * 64 + nt * 16 + rlane;
            float bv  = bias[col];
#pragma unroll
            for (int v = 0; v < 8; ++v) {
                int row = bm + wm * 32 + mt * 16 + v + 8 * half;
                __builtin_nontemporal_store((float)acc[mt][nt][v] * s + bv,
                                            out + (size_t)row * OUT_F + col);
            }
        }
    }
}

extern "C" void kernel_launch(void* const* d_in, const int* in_sizes, int n_in,
                              void* d_out, int out_size, void* d_ws, size_t ws_size,
                              hipStream_t stream) {
    const float* x        = (const float*)d_in[0];
    const float* weight   = (const float*)d_in[1];
    const float* bias     = (const float*)d_in[2];
    const float* smooth   = (const float*)d_in[3];
    const float* in_scale = (const float*)d_in[4];
    const float* w_scale  = (const float*)d_in[5];
    float*       out      = (float*)d_out;

    int8_t* wq = (int8_t*)d_ws;
    int8_t* xq = wq + (size_t)OUT_F * IN_F;

    {
        size_t thr = ((size_t)OUT_F * IN_F) / 4;
        sq_quant_w<<<dim3((unsigned)(thr / 256)), dim3(256), 0, stream>>>(weight, smooth, w_scale, wq);
    }
    {
        size_t thr = ((size_t)CHUNKS * NTOK * IN_F) / 4;
        sq_quant_x<<<dim3((unsigned)(thr / 256)), dim3(256), 0, stream>>>(x, smooth, in_scale, xq);
    }
    {
        dim3 grid(NTOK / BM, OUT_F / BN);
        sq_gemm_i8<<<grid, dim3(THREADS), 0, stream>>>(xq, wq, bias, in_scale, w_scale, out);
    }
}
